// Voice_Attention_15539191677581
// MI455X (gfx1250) — matrix operand order repaired and checked by reading, not compiled, not run on hardware
//
#include <hip/hip_runtime.h>

// ---------------- problem constants ----------------
#define B_  32
#define S_  64
#define W_  30
#define E_  300
#define VD_ 128
#define H1_ 32          // dense1 width
#define EP  304         // E padded to 19*16
#define KK  608         // 2*EP : concat K dimension (padded)
#define NEGBIG 1e12f

#define OUT_O_ELEMS (B_ * E_ * S_ * W_)      // 18,432,000 floats, then A [B,S,S]

// workspace layout (bytes)
#define WS_W2T_OFF 0
#define WS_W2T_BYTES (EP * KK * 2)           // bf16 W2^T padded [304][608]
#define WS_AM_OFF  ((WS_W2T_BYTES + 255) & ~255)

typedef __bf16 bf16_t;
typedef __attribute__((ext_vector_type(16))) __bf16 v16bf;
typedef __attribute__((ext_vector_type(8)))  __bf16 v8bf;
typedef __attribute__((ext_vector_type(4)))  __bf16 v4bf;
typedef __attribute__((ext_vector_type(8)))  float  v8f;

// ---------------- kernel 0: build padded bf16 W2^T ----------------
// W2 f32 [600][300] -> W2T bf16 [n:304][kk:608]
//   kk<300      : W2[kk][n]       (top rows -> multiply M1 = masked A@V part)
//   304<=kk<604 : W2[kk-4][n]     (bottom rows 300..599 -> multiply Z part)
//   else        : 0
__global__ void k_prep_w2t(const float* __restrict__ W2, bf16_t* __restrict__ W2T) {
  int i = blockIdx.x * 256 + threadIdx.x;
  if (i >= EP * KK) return;
  int n = i / KK, kk = i % KK;
  float v = 0.f;
  if (n < E_) {
    if (kk < E_)                       v = W2[kk * E_ + n];
    else if (kk >= EP && kk < EP + E_) v = W2[(kk - 4) * E_ + n];
  }
  W2T[i] = (bf16_t)v;
}

// ---------------- kernel 1: Q projection + attention softmax ----------------
// one block per batch, 64 threads: thread t owns query row t
__global__ __launch_bounds__(64)
void k_attn(const float* __restrict__ voice, const int* __restrict__ sent_nums,
            const float* __restrict__ W1, const float* __restrict__ b1,
            float* __restrict__ A_out, bf16_t* __restrict__ Am) {
  int b = blockIdx.x, t = threadIdx.x;
  __shared__ float sv[S_ * VD_];
  __shared__ float sq[S_ * H1_];
  for (int i = t; i < S_ * VD_; i += 64) sv[i] = voice[b * S_ * VD_ + i];
  __syncthreads();
  for (int j = 0; j < H1_; ++j) {
    float s = b1[j];
    for (int d = 0; d < VD_; ++d) s += sv[t * VD_ + d] * W1[d * H1_ + j];
    sq[t * H1_ + j] = s;
  }
  __syncthreads();
  int ns = sent_nums[b];
  float qv[H1_];
#pragma unroll
  for (int j = 0; j < H1_; ++j) qv[j] = sq[t * H1_ + j];
  float lg[S_];
  float mx = -3.4e38f;
  for (int k = 0; k < S_; ++k) {
    float s = 0.f;
#pragma unroll
    for (int j = 0; j < H1_; ++j) s += qv[j] * sq[k * H1_ + j];
    if (k >= ns) s -= NEGBIG;
    lg[k] = s;
    mx = fmaxf(mx, s);
  }
  float sum = 0.f;
  for (int k = 0; k < S_; ++k) { float e = __expf(lg[k] - mx); lg[k] = e; sum += e; }
  float inv = 1.f / sum;
  float mq = (t < ns) ? 1.f : 0.f;
  for (int k = 0; k < S_; ++k) {
    float a = lg[k] * inv;
    A_out[(b * S_ + t) * S_ + k] = a;                 // output #2 (voice_A)
    Am[(b * S_ + t) * S_ + k] = (bf16_t)(a * mq);     // row-masked bf16 A
  }
}

// ---------------- kernel 2: fused gather + (mA)@Z + concat-GEMM ----------------
// grid (W_, B_, 2): one WG per (w, b, q-half). 128 threads = 4 waves.
// LDS: sM1 [32][304] bf16 ‖ sZ [64][304] bf16  (concat along K for step 2)
__global__ __launch_bounds__(128)
void k_main(const int* __restrict__ word_idx, const float* __restrict__ emb,
            const float* __restrict__ b2, const bf16_t* __restrict__ W2T,
            const bf16_t* __restrict__ Am_all, float* __restrict__ out) {
  const int w = blockIdx.x, b = blockIdx.y, h = blockIdx.z;     // h: q-half
  const int tid = threadIdx.x, lane = tid & 31, wave = tid >> 5;
  const int hi = (lane >> 4) & 1;      // upper half-wave
  const int l16 = lane & 15;

  __shared__ __align__(16) bf16_t sArr[(32 + 64) * EP]; // 58,368 B
  bf16_t* sM1 = sArr;            // rows 0..31 (local q), cols 0..303
  bf16_t* sZ  = sArr + 32 * EP;  // rows 0..63 (k),       cols 0..303

  // ---- gather Z = emb[word_idx[b, :, w]]  (bf16, zero-padded cols 300..303)
  // vectorized: float4 global load -> v4bf (8B) LDS store
  for (int i = tid; i < S_ * (EP / 4); i += 128) {
    int s = i / (EP / 4), c = (i % (EP / 4)) * 4;
    v4bf q = {};
    if (c < E_) {
      int idx = word_idx[(b * S_ + s) * W_ + w];
      float4 f = *reinterpret_cast<const float4*>(emb + idx * E_ + c);
      q[0] = (bf16_t)f.x; q[1] = (bf16_t)f.y; q[2] = (bf16_t)f.z; q[3] = (bf16_t)f.w;
    }
    *reinterpret_cast<v4bf*>(sZ + s * EP + c) = q;
  }
  __syncthreads();

  const bf16_t* Am = Am_all + b * S_ * S_;

  // ---- phase B: M1[32,304] = (m*A)[half rows, 64] @ Z[64,304] ----
  // b-fragments hoisted and reused across the two mt tiles
  for (int nt = wave; nt < 19; nt += 4) {
    const int n = nt * 16 + l16;
    v16bf bm[2];
#pragma unroll
    for (int kt = 0; kt < 2; ++kt) {
      const int kr = kt * 32 + hi * 8;
#pragma unroll
      for (int e = 0; e < 16; ++e) bm[kt][e] = sZ[(kr + e + (e >> 3) * 8) * EP + n];
    }
#pragma unroll
    for (int mt = 0; mt < 2; ++mt) {
      const int row = h * 32 + mt * 16 + l16;   // global q row (A-operand row)
      v8f acc = {};
#pragma unroll
      for (int kt = 0; kt < 2; ++kt) {
        v8bf lo = *reinterpret_cast<const v8bf*>(Am + row * S_ + kt * 32 + hi * 8);
        v8bf hp = *reinterpret_cast<const v8bf*>(Am + row * S_ + kt * 32 + 16 + hi * 8);
        v16bf a = __builtin_shufflevector(lo, hp, 0, 1, 2, 3, 4, 5, 6, 7,
                                          8, 9, 10, 11, 12, 13, 14, 15);
        acc = __builtin_amdgcn_wmma_f32_16x16x32_bf16(false, a, false, bm[kt],
                                                      (short)0, acc, false, false);
      }
#pragma unroll
      for (int v = 0; v < 8; ++v)
        sM1[(mt * 16 + v + hi * 8) * EP + n] = (bf16_t)acc[v];
    }
  }
  __syncthreads();

  // ---- phase C: Y[32,304] = [M1 | Z_half] [32,608] @ W2T^T, + b2, store transposed
  for (int t = wave; t < 2 * 19; t += 4) {
    const int mt = t / 19, nt = t % 19;
    const int rl = mt * 16 + l16;             // local q row 0..31
    const int zrow = h * 32 + rl;             // row within sZ
    const int n = nt * 16 + l16;              // output embedding column
    v8f acc = {};
    for (int kt = 0; kt < 19; ++kt) {
      // A fragment: two 8-element contiguous K runs (each fully in one region)
      int kb0 = kt * 32 + hi * 8;
      int kb1 = kt * 32 + 16 + hi * 8;
      int off0 = (kb0 < EP) ? (rl * EP + kb0) : (32 * EP + zrow * EP + (kb0 - EP));
      int off1 = (kb1 < EP) ? (rl * EP + kb1) : (32 * EP + zrow * EP + (kb1 - EP));
      v8bf lo = *reinterpret_cast<const v8bf*>(sArr + off0);
      v8bf hp = *reinterpret_cast<const v8bf*>(sArr + off1);
      v16bf a = __builtin_shufflevector(lo, hp, 0, 1, 2, 3, 4, 5, 6, 7,
                                        8, 9, 10, 11, 12, 13, 14, 15);
      // B fragment: two 8-element K runs in one W2T row (16B aligned), same order as A
      v8bf blo = *reinterpret_cast<const v8bf*>(W2T + n * KK + kt * 32 + hi * 8);
      v8bf bhp = *reinterpret_cast<const v8bf*>(W2T + n * KK + kt * 32 + 16 + hi * 8);
      v16bf bm = __builtin_shufflevector(blo, bhp, 0, 1, 2, 3, 4, 5, 6, 7,
                                         8, 9, 10, 11, 12, 13, 14, 15);
      acc = __builtin_amdgcn_wmma_f32_16x16x32_bf16(false, a, false, bm,
                                                    (short)0, acc, false, false);
    }
    if (n < E_) {
      float bias = b2[n];
#pragma unroll
      for (int v = 0; v < 8; ++v) {
        int q = h * 32 + mt * 16 + v + hi * 8;
        // out[b, n, q, w] in [B, E, S, W]
        out[((b * E_ + n) * S_ + q) * W_ + w] = acc[v] + bias;
      }
    }
  }
}

// ---------------- host launcher ----------------
extern "C" void kernel_launch(void* const* d_in, const int* in_sizes, int n_in,
                              void* d_out, int out_size, void* d_ws, size_t ws_size,
                              hipStream_t stream) {
  (void)in_sizes; (void)n_in; (void)out_size; (void)ws_size;
  const int*   word_idx  = (const int*)d_in[0];
  const float* voice     = (const float*)d_in[1];
  const int*   sent_nums = (const int*)d_in[2];
  // d_in[3] = max_sent_nums scalar (compile-time S_)
  const float* emb_table = (const float*)d_in[4];
  const float* W1        = (const float*)d_in[5];
  const float* b1        = (const float*)d_in[6];
  const float* W2        = (const float*)d_in[7];
  const float* b2        = (const float*)d_in[8];

  float* out   = (float*)d_out;
  float* A_out = out + OUT_O_ELEMS;

  bf16_t* W2T = (bf16_t*)((char*)d_ws + WS_W2T_OFF);
  bf16_t* Am  = (bf16_t*)((char*)d_ws + WS_AM_OFF);

  int n0 = EP * KK;
  k_prep_w2t<<<(n0 + 255) / 256, 256, 0, stream>>>(W2, W2T);
  k_attn<<<B_, 64, 0, stream>>>(voice, sent_nums, W1, b1, A_out, Am);
  k_main<<<dim3(W_, B_, 2), 128, 0, stream>>>(word_idx, emb_table, b2, W2T, Am, out);
}
